// GraphTransformer_5076651344321
// MI455X (gfx1250) — matrix operand order repaired and checked by reading, not compiled, not run on hardware
//
#include <hip/hip_runtime.h>
#include <hip/hip_bf16.h>
#include <math.h>

#define NN   10000
#define EE   160000
#define IND  512
#define HID  256
#define OUTD 128
#define NH   4

typedef __attribute__((ext_vector_type(16))) _Float16 v16h;
typedef __attribute__((ext_vector_type(8)))  float    v8f;

// ---------------- elementwise helpers ----------------

__global__ void k_cvt_f32_f16(const float* __restrict__ in, _Float16* __restrict__ out, int n) {
    int i = blockIdx.x * blockDim.x + threadIdx.x;
    if (i < n) out[i] = (_Float16)in[i];
}

// W [K, Nc] fp32 row-major  ->  Wt [Nc, K] f16 (i.e. column n of W becomes row n of Wt)
__global__ void k_transpose_w(const float* __restrict__ W, _Float16* __restrict__ Wt, int K, int Nc) {
    int i = blockIdx.x * blockDim.x + threadIdx.x;
    if (i < K * Nc) {
        int k = i / Nc, n = i - k * Nc;
        Wt[(size_t)n * K + k] = (_Float16)W[i];
    }
}

__global__ void k_fill_u32(unsigned* __restrict__ p, unsigned v, int n) {
    int i = blockIdx.x * blockDim.x + threadIdx.x;
    if (i < n) p[i] = v;
}

// ---------------- WMMA GEMM: C[M,Nc] = act(A[M,K] @ W[K,Nc] + b) ----------------
// A: f16 row-major [M,K].  Wt: f16 [Nc,K] (W transposed).
// One wave computes a 16 x (16*NT) tile: the A fragment is loaded once per
// 32-wide K step and reused across NT B fragments / NT f32 accumulators.
// A frag (16x32 f16): lane L (m=L&15, half=L>>4): VGPR0-3 hold k=half*8..+7,
//   VGPR4-7 hold k=16+half*8..+7  -> two 16B contiguous loads per lane.
// B frag (32x16 f16): lane L (n=L&15, half=L>>4): same k order as A
//   -> two 16B contiguous loads per lane from Wt row n.
template <int NT>
__global__ void k_gemm_f16(const _Float16* __restrict__ A, const _Float16* __restrict__ Wt,
                           const float* __restrict__ bias, float* __restrict__ C,
                           int M, int K, int Nc, int relu, int tiles) {
    int wave = (int)((blockIdx.x * blockDim.x + threadIdx.x) >> 5);
    if (wave >= tiles) return;                 // wave-uniform: EXEC stays all-ones for WMMA
    int lane   = threadIdx.x & 31;
    int tilesN = Nc / (16 * NT);
    int tm = wave / tilesN, tn0 = (wave - tm * tilesN) * NT;
    int half = lane >> 4, r = lane & 15;

    const _Float16* arow  = A  + (size_t)(tm * 16 + r) * K;
    const _Float16* bbase = Wt + (size_t)(tn0 * 16 + r) * K;

    v8f acc[NT];
#pragma unroll
    for (int t = 0; t < NT; ++t) acc[t] = (v8f){};

    for (int kb = 0; kb < K; kb += 32) {
        union { v16h h; uint4 q[2]; } a;
        a.q[0] = *(const uint4*)(arow + kb + half * 8);
        a.q[1] = *(const uint4*)(arow + kb + 16 + half * 8);
        union { v16h h; uint4 q[2]; } b[NT];
#pragma unroll
        for (int t = 0; t < NT; ++t) {
            const _Float16* brow = bbase + (size_t)t * 16 * K;
            b[t].q[0] = *(const uint4*)(brow + kb + half * 8);
            b[t].q[1] = *(const uint4*)(brow + kb + 16 + half * 8);
        }
#pragma unroll
        for (int t = 0; t < NT; ++t)
            acc[t] = __builtin_amdgcn_wmma_f32_16x16x32_f16(false, a.h, false, b[t].h,
                                                            (short)0, acc[t], false, false);
    }
    // C/D layout: VGPR v -> row m = v + half*8, lane -> col n = r
#pragma unroll
    for (int t = 0; t < NT; ++t) {
        int   n  = (tn0 + t) * 16 + r;
        float bn = bias[n];
        float* crow = C + (size_t)(tm * 16 + half * 8) * Nc + n;
#pragma unroll
        for (int v = 0; v < 8; ++v) {
            float val = acc[t][v] + bn;
            if (relu) val = fmaxf(val, 0.0f);
            crow[(size_t)v * Nc] = val;
        }
    }
}

// ---------------- edge attention ----------------

__device__ __forceinline__ unsigned fkey(float f) {   // order-preserving uint key
    unsigned u = __float_as_uint(f);
    return (u & 0x80000000u) ? ~u : (u | 0x80000000u);
}
__device__ __forceinline__ float finv(unsigned u) {
    return (u & 0x80000000u) ? __uint_as_float(u & 0x7FFFFFFFu) : __uint_as_float(~u);
}

__global__ void k_edge_logits(const float* __restrict__ Q, const float* __restrict__ Kf,
                              const int* __restrict__ src, const int* __restrict__ dst,
                              float* __restrict__ logits, int C, int dh, float scale) {
    int i = blockIdx.x * blockDim.x + threadIdx.x;
    if (i >= EE * NH) return;
    int e = i >> 2, h = i & 3;
    int s = src[e], d = dst[e];
    const float* qp = Q  + (size_t)d * C + h * dh;
    const float* kp = Kf + (size_t)s * C + h * dh;
    float acc = 0.f;
    for (int j = 0; j < dh; ++j) acc += qp[j] * kp[j];
    logits[i] = acc * scale;
}

__global__ void k_seg_max(const float* __restrict__ logits, const int* __restrict__ dst,
                          unsigned* __restrict__ mbits) {
    int i = blockIdx.x * blockDim.x + threadIdx.x;
    if (i >= EE * NH) return;
    int e = i >> 2, h = i & 3;
    atomicMax(&mbits[dst[e] * NH + h], fkey(logits[i]));
}

__global__ void k_edge_exp(float* __restrict__ logits, const int* __restrict__ dst,
                           const unsigned* __restrict__ mbits, float* __restrict__ denom) {
    int i = blockIdx.x * blockDim.x + threadIdx.x;
    if (i >= EE * NH) return;
    int e = i >> 2, h = i & 3;
    int d = dst[e];
    float m = finv(mbits[d * NH + h]);
    float w = expf(logits[i] - m);
    logits[i] = w;
    atomicAdd(&denom[d * NH + h], w);
}

__global__ void k_edge_agg(const float* __restrict__ w, const float* __restrict__ denom,
                           const float* __restrict__ V, const int* __restrict__ src,
                           const int* __restrict__ dst, float* __restrict__ agg,
                           int C, int dh) {
    int i = blockIdx.x * blockDim.x + threadIdx.x;
    if (i >= EE * NH) return;
    int e = i >> 2, h = i & 3;
    int s = src[e], d = dst[e];
    float alpha = w[i] / denom[d * NH + h];
    const float* vp = V   + (size_t)s * C + h * dh;
    float*       ap = agg + (size_t)d * C + h * dh;
    for (int j = 0; j < dh; ++j) atomicAdd(&ap[j], alpha * vp[j]);
}

// ---------------- LayerNorm(agg + skip) -> f16, one wave per node ----------------

__global__ void k_add_ln(const float* __restrict__ agg, const float* __restrict__ skip,
                         const float* __restrict__ g, const float* __restrict__ b,
                         _Float16* __restrict__ out16, int C, int relu) {
    int node = (int)((blockIdx.x * blockDim.x + threadIdx.x) >> 5);
    if (node >= NN) return;
    int lane = threadIdx.x & 31;
    int per  = C >> 5;                 // C=256 -> 8, C=128 -> 4
    const float* ap = agg  + (size_t)node * C;
    const float* sp = skip + (size_t)node * C;
    float vals[8];
    float sum = 0.f;
    for (int j = 0; j < per; ++j) {
        float v = ap[lane + j * 32] + sp[lane + j * 32];
        vals[j] = v; sum += v;
    }
    for (int off = 16; off > 0; off >>= 1) sum += __shfl_xor(sum, off, 32);
    float mu = sum / (float)C;
    float var = 0.f;
    for (int j = 0; j < per; ++j) { float d = vals[j] - mu; var += d * d; }
    for (int off = 16; off > 0; off >>= 1) var += __shfl_xor(var, off, 32);
    float inv = rsqrtf(var / (float)C + 1e-5f);
    for (int j = 0; j < per; ++j) {
        int c = lane + j * 32;
        float v = (vals[j] - mu) * inv * g[c] + b[c];
        if (relu) v = fmaxf(v, 0.f);
        out16[(size_t)node * C + c] = (_Float16)v;
    }
}

// ---------------- final row L2-normalize, one wave per node ----------------

__global__ void k_normalize(const float* __restrict__ Y, float* __restrict__ out) {
    int node = (int)((blockIdx.x * blockDim.x + threadIdx.x) >> 5);
    if (node >= NN) return;
    int lane = threadIdx.x & 31;
    const float* yp = Y + (size_t)node * OUTD;
    float vals[4];
    float ss = 0.f;
    for (int j = 0; j < 4; ++j) { float v = yp[lane + j * 32]; vals[j] = v; ss += v * v; }
    for (int off = 16; off > 0; off >>= 1) ss += __shfl_xor(ss, off, 32);
    float inv = 1.f / fmaxf(sqrtf(ss), 1e-12f);
    for (int j = 0; j < 4; ++j) out[(size_t)node * OUTD + lane + j * 32] = vals[j] * inv;
}

// ---------------- host orchestration ----------------

extern "C" void kernel_launch(void* const* d_in, const int* in_sizes, int n_in,
                              void* d_out, int out_size, void* d_ws, size_t ws_size,
                              hipStream_t stream) {
    (void)out_size; (void)ws_size;

    // Locate x and edge_index by their unique sizes; remaining leaves are the
    // params pytree in JAX tree_leaves (alphabetical-key) order.
    int ix = -1, ie = -1;
    for (int i = 0; i < n_in; ++i) {
        if (in_sizes[i] == NN * IND && ix < 0) ix = i;
        else if (in_sizes[i] == 2 * EE && ie < 0) ie = i;
    }
    if (ix < 0 || ie < 0) return;
    const float* pl[24]; int np = 0;
    for (int i = 0; i < n_in && np < 24; ++i)
        if (i != ix && i != ie) pl[np++] = (const float*)d_in[i];
    if (np < 24) return;

    const float* x    = (const float*)d_in[ix];
    const int*   eidx = (const int*)d_in[ie];
    const int*   srcv = eidx;
    const int*   dstv = eidx + EE;

    // params (sorted keys): input_proj{b,w}, layers[0]{k{b,w},ln_b,ln_g,q{b,w},skip{b,w},v{b,w}},
    //                       layers[1]{...}, output_proj{b,w}
    const float *ip_b = pl[0],  *ip_w = pl[1];
    const float *l0kb = pl[2],  *l0kw = pl[3],  *l0lnb = pl[4],  *l0lng = pl[5];
    const float *l0qb = pl[6],  *l0qw = pl[7],  *l0sb  = pl[8],  *l0sw  = pl[9];
    const float *l0vb = pl[10], *l0vw = pl[11];
    const float *l1kb = pl[12], *l1kw = pl[13], *l1lnb = pl[14], *l1lng = pl[15];
    const float *l1qb = pl[16], *l1qw = pl[17], *l1sb  = pl[18], *l1sw  = pl[19];
    const float *l1vb = pl[20], *l1vw = pl[21];
    const float *ob   = pl[22], *ow   = pl[23];

    // workspace carve-up
    char* ws = (char*)d_ws;
    size_t off = 0;
    auto alloc = [&](size_t bytes) -> char* {
        char* p = ws + off;
        off += (bytes + 255) & ~(size_t)255;
        return p;
    };
    _Float16* Xh    = (_Float16*)alloc((size_t)NN * IND * 2);
    _Float16* WtIn  = (_Float16*)alloc((size_t)IND * HID * 2);
    _Float16* WtQ0  = (_Float16*)alloc((size_t)HID * HID * 2);
    _Float16* WtK0  = (_Float16*)alloc((size_t)HID * HID * 2);
    _Float16* WtV0  = (_Float16*)alloc((size_t)HID * HID * 2);
    _Float16* WtS0  = (_Float16*)alloc((size_t)HID * HID * 2);
    _Float16* WtQ1  = (_Float16*)alloc((size_t)HID * OUTD * 2);
    _Float16* WtK1  = (_Float16*)alloc((size_t)HID * OUTD * 2);
    _Float16* WtV1  = (_Float16*)alloc((size_t)HID * OUTD * 2);
    _Float16* WtS1  = (_Float16*)alloc((size_t)HID * OUTD * 2);
    _Float16* WtOut = (_Float16*)alloc((size_t)OUTD * OUTD * 2);
    _Float16* H0h   = (_Float16*)alloc((size_t)NN * HID * 2);
    float*    Qb    = (float*)alloc((size_t)NN * HID * 4);   // reused by layer 1 (OUTD<HID)
    float*    Kb    = (float*)alloc((size_t)NN * HID * 4);
    float*    Vb    = (float*)alloc((size_t)NN * HID * 4);
    float*    Sb    = (float*)alloc((size_t)NN * HID * 4);
    float*    logit = (float*)alloc((size_t)EE * NH * 4);
    unsigned* mbits = (unsigned*)alloc((size_t)NN * NH * 4);
    float*    denom = (float*)alloc((size_t)NN * NH * 4);
    float*    agg   = (float*)alloc((size_t)NN * HID * 4);   // also tmp f32 GEMM out, reused by layer 1
    _Float16* H1h   = (_Float16*)alloc((size_t)NN * HID * 2);
    _Float16* H2h   = (_Float16*)alloc((size_t)NN * OUTD * 2);
    float*    Y     = (float*)alloc((size_t)NN * OUTD * 4);

    const int TB = 256;
    auto nb = [](long n, int per) { return (int)((n + per - 1) / per); };

    // stage 0: conversions
    k_cvt_f32_f16<<<nb((long)NN * IND, TB), TB, 0, stream>>>(x, Xh, NN * IND);
    k_transpose_w<<<nb((long)IND * HID, TB),  TB, 0, stream>>>(ip_w, WtIn, IND, HID);
    k_transpose_w<<<nb((long)HID * HID, TB),  TB, 0, stream>>>(l0qw, WtQ0, HID, HID);
    k_transpose_w<<<nb((long)HID * HID, TB),  TB, 0, stream>>>(l0kw, WtK0, HID, HID);
    k_transpose_w<<<nb((long)HID * HID, TB),  TB, 0, stream>>>(l0vw, WtV0, HID, HID);
    k_transpose_w<<<nb((long)HID * HID, TB),  TB, 0, stream>>>(l0sw, WtS0, HID, HID);
    k_transpose_w<<<nb((long)HID * OUTD, TB), TB, 0, stream>>>(l1qw, WtQ1, HID, OUTD);
    k_transpose_w<<<nb((long)HID * OUTD, TB), TB, 0, stream>>>(l1kw, WtK1, HID, OUTD);
    k_transpose_w<<<nb((long)HID * OUTD, TB), TB, 0, stream>>>(l1vw, WtV1, HID, OUTD);
    k_transpose_w<<<nb((long)HID * OUTD, TB), TB, 0, stream>>>(l1sw, WtS1, HID, OUTD);
    k_transpose_w<<<nb((long)OUTD * OUTD, TB), TB, 0, stream>>>(ow, WtOut, OUTD, OUTD);

    // 16 x 64 tile per wave (NT=4)
    const int tilesH = (NN / 16) * (HID / 64);   // 625*4 = 2500
    const int tilesO = (NN / 16) * (OUTD / 64);  // 625*2 = 1250
    const int gH = (tilesH + 7) / 8;             // 8 waves per 256-thread block
    const int gO = (tilesO + 7) / 8;

    // stage 1: h = relu(x @ Win + b)  (agg used as f32 temp, then cvt to f16)
    k_gemm_f16<4><<<gH, TB, 0, stream>>>(Xh, WtIn, ip_b, agg, NN, IND, HID, 1, tilesH);
    k_cvt_f32_f16<<<nb((long)NN * HID, TB), TB, 0, stream>>>(agg, H0h, NN * HID);

    // ---- layer 0 ----
    k_gemm_f16<4><<<gH, TB, 0, stream>>>(H0h, WtQ0, l0qb, Qb, NN, HID, HID, 0, tilesH);
    k_gemm_f16<4><<<gH, TB, 0, stream>>>(H0h, WtK0, l0kb, Kb, NN, HID, HID, 0, tilesH);
    k_gemm_f16<4><<<gH, TB, 0, stream>>>(H0h, WtV0, l0vb, Vb, NN, HID, HID, 0, tilesH);
    k_gemm_f16<4><<<gH, TB, 0, stream>>>(H0h, WtS0, l0sb, Sb, NN, HID, HID, 0, tilesH);
    k_fill_u32<<<nb((long)NN * NH, TB),  TB, 0, stream>>>(mbits, 0u, NN * NH);
    k_fill_u32<<<nb((long)NN * NH, TB),  TB, 0, stream>>>((unsigned*)denom, 0u, NN * NH);
    k_fill_u32<<<nb((long)NN * HID, TB), TB, 0, stream>>>((unsigned*)agg, 0u, NN * HID);
    k_edge_logits<<<nb((long)EE * NH, TB), TB, 0, stream>>>(Qb, Kb, srcv, dstv, logit,
                                                            HID, HID / NH, 0.125f); // 1/sqrt(64)
    k_seg_max<<<nb((long)EE * NH, TB), TB, 0, stream>>>(logit, dstv, mbits);
    k_edge_exp<<<nb((long)EE * NH, TB), TB, 0, stream>>>(logit, dstv, mbits, denom);
    k_edge_agg<<<nb((long)EE * NH, TB), TB, 0, stream>>>(logit, denom, Vb, srcv, dstv,
                                                         agg, HID, HID / NH);
    k_add_ln<<<NN * 32 / TB, TB, 0, stream>>>(agg, Sb, l0lng, l0lnb, H1h, HID, 1);

    // ---- layer 1 ----
    k_gemm_f16<4><<<gO, TB, 0, stream>>>(H1h, WtQ1, l1qb, Qb, NN, HID, OUTD, 0, tilesO);
    k_gemm_f16<4><<<gO, TB, 0, stream>>>(H1h, WtK1, l1kb, Kb, NN, HID, OUTD, 0, tilesO);
    k_gemm_f16<4><<<gO, TB, 0, stream>>>(H1h, WtV1, l1vb, Vb, NN, HID, OUTD, 0, tilesO);
    k_gemm_f16<4><<<gO, TB, 0, stream>>>(H1h, WtS1, l1sb, Sb, NN, HID, OUTD, 0, tilesO);
    k_fill_u32<<<nb((long)NN * NH, TB),   TB, 0, stream>>>(mbits, 0u, NN * NH);
    k_fill_u32<<<nb((long)NN * NH, TB),   TB, 0, stream>>>((unsigned*)denom, 0u, NN * NH);
    k_fill_u32<<<nb((long)NN * OUTD, TB), TB, 0, stream>>>((unsigned*)agg, 0u, NN * OUTD);
    k_edge_logits<<<nb((long)EE * NH, TB), TB, 0, stream>>>(Qb, Kb, srcv, dstv, logit,
                                                            OUTD, OUTD / NH, 0.1767766953f); // 1/sqrt(32)
    k_seg_max<<<nb((long)EE * NH, TB), TB, 0, stream>>>(logit, dstv, mbits);
    k_edge_exp<<<nb((long)EE * NH, TB), TB, 0, stream>>>(logit, dstv, mbits, denom);
    k_edge_agg<<<nb((long)EE * NH, TB), TB, 0, stream>>>(logit, denom, Vb, srcv, dstv,
                                                         agg, OUTD, OUTD / NH);
    k_add_ln<<<NN * 32 / TB, TB, 0, stream>>>(agg, Sb, l1lng, l1lnb, H2h, OUTD, 0);

    // ---- output projection + L2 normalize ----
    k_gemm_f16<4><<<gO, TB, 0, stream>>>(H2h, WtOut, ob, Y, NN, OUTD, OUTD, 0, tilesO);
    k_normalize<<<NN * 32 / TB, TB, 0, stream>>>(Y, (float*)d_out);
}
